// MultiBandLinearGenerator_31653908971998
// MI455X (gfx1250) — matrix operand order repaired and checked by reading, not compiled, not run on hardware
//
#include <hip/hip_runtime.h>

typedef _Float16 v16h __attribute__((ext_vector_type(16)));
typedef _Float16 v8h  __attribute__((ext_vector_type(8)));
typedef float    v8f  __attribute__((ext_vector_type(8)));

#define IN_CH 5
#define MID   32
#define IMG   512
#define OIMG  256
#define CT    64            // conv-output tile edge per block
#define IT    (CT + 12)     // 76: input tile edge (halo 6 each side)
#define ISTR  80            // input row stride in halfs (16B aligned, >= 80 cols touched)
#define KD    13            // effective kernel edge
#define TBFRAG (KD * 32 * 16)   // Toeplitz frag halfs per channel

// ---------------------------------------------------------------------------
// Kernel 1: collapse the 6-conv linear chain into a single 13x13 kernel per
// channel, then emit WMMA B-operand Toeplitz fragments (exact lane striping).
// ---------------------------------------------------------------------------
__global__ __launch_bounds__(256)
void build_kernels(const float* __restrict__ W0, const float* __restrict__ W1,
                   const float* __restrict__ W2, const float* __restrict__ W3,
                   const float* __restrict__ W4, const float* __restrict__ W5,
                   _Float16* __restrict__ tb) {
  const int c   = blockIdx.x;
  const int tid = threadIdx.x;

  __shared__ float W0s[MID * 49];    // 7x7, 1->32
  __shared__ float E1[MID * 121];    // 11x11, 1->32  (W0 * W1)
  __shared__ float E2[MID * 169];    // 13x13, 1->32  (E1 * W2)
  __shared__ float t5j[MID], wvec[MID], Kc[169];

  for (int i = tid; i < MID * 49; i += 256) W0s[i] = W0[c * MID * 49 + i];
  __syncthreads();

  // E1[oc][u] = sum_ic sum_{s+t=u} W1[c][oc][ic][s] * W0[c][ic][t]
  for (int i = tid; i < MID * 121; i += 256) {
    int oc = i / 121, u = i - oc * 121, uy = u / 11, ux = u - uy * 11;
    float s = 0.f;
    for (int ic = 0; ic < MID; ++ic) {
      const float* w1 = W1 + (((size_t)c * MID + oc) * MID + ic) * 25;
      int sy0 = uy > 6 ? uy - 6 : 0, sy1 = uy < 4 ? uy : 4;
      int sx0 = ux > 6 ? ux - 6 : 0, sx1 = ux < 4 ? ux : 4;
      for (int sy = sy0; sy <= sy1; ++sy)
        for (int sx = sx0; sx <= sx1; ++sx)
          s += w1[sy * 5 + sx] * W0s[ic * 49 + (uy - sy) * 7 + (ux - sx)];
    }
    E1[i] = s;
  }
  __syncthreads();

  // E2[oc][u] = sum_ic sum_{s+t=u} W2[c][oc][ic][s] * E1[ic][t]
  for (int i = tid; i < MID * 169; i += 256) {
    int oc = i / 169, u = i - oc * 169, uy = u / 13, ux = u - uy * 13;
    float s = 0.f;
    for (int ic = 0; ic < MID; ++ic) {
      const float* w2 = W2 + (((size_t)c * MID + oc) * MID + ic) * 9;
      int sy0 = uy > 10 ? uy - 10 : 0, sy1 = uy < 2 ? uy : 2;
      int sx0 = ux > 10 ? ux - 10 : 0, sx1 = ux < 2 ? ux : 2;
      for (int sy = sy0; sy <= sy1; ++sy)
        for (int sx = sx0; sx <= sx1; ++sx)
          s += w2[sy * 3 + sx] * E1[ic * 121 + (uy - sy) * 11 + (ux - sx)];
    }
    E2[i] = s;
  }
  // 1x1 chain: wvec[ic] = sum_j (sum_k W5[0][k] W4[k][j]) W3[j][ic]
  if (tid < MID) {
    float s = 0.f;
    for (int k = 0; k < MID; ++k)
      s += W5[c * MID + k] * W4[((size_t)c * MID + k) * MID + tid];
    t5j[tid] = s;
  }
  __syncthreads();
  if (tid < MID) {
    float s = 0.f;
    for (int j = 0; j < MID; ++j)
      s += t5j[j] * W3[((size_t)c * MID + j) * MID + tid];
    wvec[tid] = s;
  }
  __syncthreads();

  // Kc[u] = sum_oc wvec[oc] * E2[oc][u]   (13x13 effective kernel)
  for (int i = tid; i < 169; i += 256) {
    float s = 0.f;
    for (int oc = 0; oc < MID; ++oc) s += wvec[oc] * E2[oc * 169 + i];
    Kc[i] = s;
  }
  __syncthreads();

  // Toeplitz B fragments: tb[c][dy][lane][e] = Kc[dy][K - x'] (0 if OOR),
  // K = e + 16*(lane>=16), x' = lane%16  -> contiguous k order, same as the A fragment.
  for (int i = tid; i < TBFRAG; i += 256) {
    int dy = i / 512, L = (i >> 4) & 31, e = i & 15;
    int K  = e + ((L >= 16) ? 16 : 0);
    int dx = K - (L & 15);
    float v = (dx >= 0 && dx < KD) ? Kc[dy * KD + dx] : 0.f;
    tb[(size_t)c * TBFRAG + i] = (_Float16)v;
  }
}

// ---------------------------------------------------------------------------
// Kernel 2: per (b,c) 64x64 conv tile via 16x16 WMMA sub-tiles (13 x
// v_wmma_f32_16x16x32_f16 each), fused 2x2 mean pool, f32 output.
// ---------------------------------------------------------------------------
__global__ __launch_bounds__(128)
void fused_conv(const float* __restrict__ x, const _Float16* __restrict__ tb,
                float* __restrict__ out) {
  __shared__ _Float16 inBuf[IT * ISTR];   // 76*80 f16 = 12160 B
  __shared__ float    convOut[CT * CT];   // 16384 B

  const int tid = threadIdx.x;
  const int bz  = blockIdx.z;
  const int b   = bz / IN_CH, c = bz % IN_CH;
  const int iy0 = blockIdx.y * CT - 6;
  const int ix0 = blockIdx.x * CT - 6;
  const float* img = x + (size_t)(b * IN_CH + c) * IMG * IMG;

  // Stage input tile (f32 -> f16), zero-pad borders and stride padding.
  for (int i = tid; i < IT * ISTR; i += 128) {
    int r = i / ISTR, col = i - r * ISTR;
    float v = 0.f;
    int gy = iy0 + r, gx = ix0 + col;
    if (col < IT && (unsigned)gy < IMG && (unsigned)gx < IMG)
      v = img[gy * IMG + gx];
    inBuf[i] = (_Float16)v;
  }
  __syncthreads();

  const int wave = tid >> 5, lane = tid & 31;
  const int lrow = lane & 15;
  const int koff = (lane < 16) ? 0 : 16;           // A-matrix K striping
  const _Float16* tbc = tb + (size_t)c * TBFRAG + lane * 16;

  for (int st = wave; st < 16; st += 4) {          // 4x4 sub-tiles, 4 waves
    const int y0 = (st >> 2) << 4;
    const int x0 = (st & 3) << 4;
    v8f acc = {0.f, 0.f, 0.f, 0.f, 0.f, 0.f, 0.f, 0.f};
    const _Float16* arow = inBuf + (y0 + lrow) * ISTR + x0 + koff;
#pragma unroll
    for (int dy = 0; dy < KD; ++dy) {
      v16h bfrag = *(const v16h*)(tbc + dy * 512);       // 32B, global (L0-hot)
      v8h alo = *(const v8h*)(arow);                     // K{0..7}/{16..23}
      v8h ahi = *(const v8h*)(arow + 8);                 // K{8..15}/{24..31}
      v16h afrag;
#pragma unroll
      for (int e = 0; e < 8; ++e) { afrag[e] = alo[e]; afrag[e + 8] = ahi[e]; }
      acc = __builtin_amdgcn_wmma_f32_16x16x32_f16(
          /*neg_a=*/false, afrag, /*neg_b=*/false, bfrag,
          /*c_mod=*/(short)0, acc, /*reuse_a=*/false, /*reuse_b=*/false);
      arow += ISTR;
    }
    // D layout: VGPR r -> M=r (lanes 0-15) / M=r+8 (lanes 16-31); N = lane%16
    const int cy = y0 + ((lane < 16) ? 0 : 8);
    const int cx = x0 + lrow;
#pragma unroll
    for (int r = 0; r < 8; ++r) convOut[(cy + r) * CT + cx] = acc[r];
  }
  __syncthreads();

  // Fused 2x2 mean pool -> (B,5,256,256) f32
  const int HT = CT / 2;
  float* obase = out + (size_t)(b * IN_CH + c) * OIMG * OIMG;
  const int oy0 = blockIdx.y * HT, ox0 = blockIdx.x * HT;
  for (int i = tid; i < HT * HT; i += 128) {
    int py = i / HT, px = i - py * HT;
    const float* q = convOut + (2 * py) * CT + 2 * px;
    float s = 0.25f * (q[0] + q[1] + q[CT] + q[CT + 1]);
    obase[(oy0 + py) * OIMG + ox0 + px] = s;
  }
}

// ---------------------------------------------------------------------------
extern "C" void kernel_launch(void* const* d_in, const int* in_sizes, int n_in,
                              void* d_out, int out_size, void* d_ws, size_t ws_size,
                              hipStream_t stream) {
  const float* x  = (const float*)d_in[0];
  const float* W0 = (const float*)d_in[1];
  const float* W1 = (const float*)d_in[2];
  const float* W2 = (const float*)d_in[3];
  const float* W3 = (const float*)d_in[4];
  const float* W4 = (const float*)d_in[5];
  const float* W5 = (const float*)d_in[6];
  float* out = (float*)d_out;
  _Float16* tb = (_Float16*)d_ws;   // 5 * 13 * 512 halfs = 66560 B

  const int Bn = in_sizes[0] / (IN_CH * IMG * IMG);   // = 8

  build_kernels<<<dim3(IN_CH), dim3(256), 0, stream>>>(W0, W1, W2, W3, W4, W5, tb);
  fused_conv<<<dim3(IMG / CT, IMG / CT, Bn * IN_CH), dim3(128), 0, stream>>>(x, tb, out);
}
